// SelfAttn_71184787964285
// MI455X (gfx1250) — matrix operand order repaired and checked by reading, compiled, not run on hardware
//
#include <hip/hip_runtime.h>

// ---------------------------------------------------------------- types ----
typedef __bf16 bf16;
typedef __attribute__((ext_vector_type(16))) __bf16 v16bf;
typedef __attribute__((ext_vector_type(8)))  __bf16 v8bf;
typedef __attribute__((ext_vector_type(8)))  float  v8f;

static constexpr int BB  = 4;
static constexpr int SS  = 2048;
static constexpr int DD  = 1024;
static constexpr int NHH = 16;
static constexpr int AD  = 64;

// ---- TDM availability -------------------------------------------------------
#if defined(__has_builtin)
#  if __has_builtin(__builtin_amdgcn_tensor_load_to_lds)
#    define HAVE_TDM 1
#  endif
#endif
#ifndef HAVE_TDM
#  define HAVE_TDM 0
#endif

#if HAVE_TDM
typedef unsigned int u32x4 __attribute__((ext_vector_type(4)));
typedef int          i32x4 __attribute__((ext_vector_type(4)));
typedef int          i32x8 __attribute__((ext_vector_type(8)));

// Issue a 2D tile DMA global->LDS via the Tensor Data Mover.
// Elements are 2 bytes (bf16). LDS rows get 4 dwords of padding after every
// 32 dwords (i.e. row stride 72 bf16 for a 64-element row).
__device__ __forceinline__ void tdm_load_2d(unsigned lds_off, const void* gaddr,
                                            unsigned tensor_d0, unsigned tensor_d1,
                                            unsigned tile_d0, unsigned tile_d1,
                                            unsigned stride0) {
    unsigned long long ga = (unsigned long long)(size_t)gaddr;
    u32x4 g0;
    g0[0] = 1u;                                           // count=1 (valid, user)
    g0[1] = lds_off;                                      // lds_addr (bytes)
    g0[2] = (unsigned)(ga & 0xffffffffu);                 // global_addr lo
    g0[3] = (unsigned)((ga >> 32) & 0x01ffffffu)          // global_addr hi
          | (2u << 30);                                   // type=2 (image/D#)
    i32x8 g1;
    g1[0] = (int)((1u << 16)                              // data_size = 2 bytes
                | (1u << 20)                              // pad_enable
                | (4u << 22)                              // pad_interval: 32 dwords
                | (3u << 25));                            // pad_amount: 4 dwords
    g1[1] = (int)((tensor_d0 & 0xffffu) << 16);           // tensor_dim0[15:0]
    g1[2] = (int)(((tensor_d0 >> 16) & 0xffffu)           // tensor_dim0[31:16]
                | ((tensor_d1 & 0xffffu) << 16));         // tensor_dim1[15:0]
    g1[3] = (int)(((tensor_d1 >> 16) & 0xffffu)           // tensor_dim1[31:16]
                | ((tile_d0 & 0xffffu) << 16));           // tile_dim0
    g1[4] = (int)(tile_d1 & 0xffffu);                     // tile_dim1, tile_dim2=0
    g1[5] = (int)stride0;                                 // tensor_dim0_stride lo
    g1[6] = 0;                                            // stride0 hi / stride1 lo
    g1[7] = 0;                                            // stride1 hi
    i32x4 z4; z4[0] = 0; z4[1] = 0; z4[2] = 0; z4[3] = 0; // 2D: groups 2/3 unused
#if defined(__clang_major__) && (__clang_major__ >= 23)
    i32x8 z8; 
    #pragma unroll
    for (int i = 0; i < 8; ++i) z8[i] = 0;
    __builtin_amdgcn_tensor_load_to_lds(g0, g1, z4, z4, z8, 0);
#else
    __builtin_amdgcn_tensor_load_to_lds(g0, g1, z4, z4, 0);
#endif
}
#endif // HAVE_TDM

// fp32 -> bf16, round-to-nearest-even
__device__ __forceinline__ bf16 f2bf(float x) {
    union { float f; unsigned u; } c; c.f = x;
    unsigned r = c.u + 0x7fffu + ((c.u >> 16) & 1u);
    union { unsigned short s; bf16 b; } o; o.s = (unsigned short)(r >> 16);
    return o.b;
}

__device__ __forceinline__ v8f vzero8() {
    v8f z;
    #pragma unroll
    for (int i = 0; i < 8; ++i) z[i] = 0.f;
    return z;
}

__device__ __forceinline__ v8f wmma_bf16(v16bf a, v16bf b, v8f c) {
    // D = A(16x32 bf16) x B(32x16 bf16) + C(16x16 f32)
    return __builtin_amdgcn_wmma_f32_16x16x32_bf16(
        /*neg_a=*/false, a, /*neg_b=*/false, b,
        /*c_mod=*/(short)0, c, /*reuse_a=*/false, /*reuse_b=*/false);
}

// A-fragment (16x32, M=row, K contiguous in memory), row-major base[ld] bf16.
// Lane layout (ISA 7.12.2): lanes 0-15 M=lane K in {0..7,16..23},
//                           lanes 16-31 M=lane-16 K in {8..15,24..31}.
__device__ __forceinline__ v16bf load_afrag(const bf16* base, int ld) {
    const int lane = threadIdx.x & 31;
    const int m = lane & 15, hl = lane >> 4;
    union { v16bf f; v8bf h[2]; } u;
    u.h[0] = *(const v8bf*)(base + (size_t)m * ld + hl * 8);
    u.h[1] = *(const v8bf*)(base + (size_t)m * ld + 16 + hl * 8);
    return u.f;
}

// B-fragment (32x16, N=row of storage, K contiguous): lanes 0-15 N=lane K in {0..7,16..23},
// lanes 16-31 N=lane-16 K in {8..15,24..31}. base points at [n0][k0] of an n-major tile.
__device__ __forceinline__ v16bf load_bfrag(const bf16* base, int ld) {
    const int lane = threadIdx.x & 31;
    const int n = lane & 15, hl = lane >> 4;
    union { v16bf f; v8bf h[2]; } u;
    const bf16* p = base + (size_t)n * ld + hl * 8;
    u.h[0] = *(const v8bf*)(p);
    u.h[1] = *(const v8bf*)(p + 16);
    return u.f;
}

// ------------------------------------------------- kernel 1: weight convert
__global__ void cvt_bf16_kernel(const float* __restrict__ src,
                                bf16* __restrict__ dst, int n) {
    int i = blockIdx.x * blockDim.x + threadIdx.x;
    int stride = gridDim.x * blockDim.x;
    for (; i < n; i += stride) dst[i] = f2bf(src[i]);
}

// ------------------------------------- kernel 2: QKV GEMM + Q/K/Vt scatter
// X:(B*S, D) f32, W:(3D, D) bf16 -> Q,K:[b,h,s,a] bf16, Vt:[b,h,a,s] bf16
__global__ __launch_bounds__(256)
void qkv_gemm_kernel(const float* __restrict__ X, const bf16* __restrict__ W,
                     bf16* __restrict__ Qb, bf16* __restrict__ Kb,
                     bf16* __restrict__ Vtb) {
    __shared__ bf16 As[128][40];
    __shared__ bf16 Bs[128][40];
    const int tid = threadIdx.x, lane = tid & 31, w = tid >> 5;
    const int wm = w >> 1, wn = w & 1;           // 4x2 wave grid, 32x64 tiles
    const int m0 = blockIdx.x * 128;
    const int n0 = blockIdx.y * 128;
    const int hl = lane >> 4, nn = lane & 15;

    v8f acc[2][4];
    #pragma unroll
    for (int i = 0; i < 2; ++i)
        #pragma unroll
        for (int j = 0; j < 4; ++j) acc[i][j] = vzero8();

    for (int k0 = 0; k0 < DD; k0 += 32) {
        // stage A: 128x32 f32 -> bf16 (and prefetch next k-tile)
        for (int r = tid >> 3; r < 128; r += 32) {
            int c = (tid & 7) * 4;
            const float* px = X + (size_t)(m0 + r) * DD + k0 + c;
            const float4 v = *(const float4*)px;
            if (k0 + 32 < DD) __builtin_prefetch(px + 32, 0, 1);
            As[r][c + 0] = f2bf(v.x); As[r][c + 1] = f2bf(v.y);
            As[r][c + 2] = f2bf(v.z); As[r][c + 3] = f2bf(v.w);
        }
        // stage B: 128x32 bf16
        for (int r = tid >> 2; r < 128; r += 64) {
            int c = (tid & 3) * 8;
            const bf16* pw = W + (size_t)(n0 + r) * DD + k0 + c;
            *(v8bf*)&Bs[r][c] = *(const v8bf*)pw;
            if (k0 + 32 < DD) __builtin_prefetch(pw + 32, 0, 1);
        }
        __syncthreads();
        v16bf af[2], bfr[4];
        af[0] = load_afrag(&As[wm * 32 + 0][0], 40);
        af[1] = load_afrag(&As[wm * 32 + 16][0], 40);
        #pragma unroll
        for (int j = 0; j < 4; ++j) bfr[j] = load_bfrag(&Bs[wn * 64 + j * 16][0], 40);
        #pragma unroll
        for (int i = 0; i < 2; ++i)
            #pragma unroll
            for (int j = 0; j < 4; ++j)
                acc[i][j] = wmma_bf16(af[i], bfr[j], acc[i][j]);
        __syncthreads();
    }

    // epilogue: scatter into Q / K / V-transposed (bf16)
    #pragma unroll
    for (int i = 0; i < 2; ++i)
        #pragma unroll
        for (int j = 0; j < 4; ++j)
            #pragma unroll
            for (int r = 0; r < 8; ++r) {
                int row = m0 + wm * 32 + i * 16 + r + 8 * hl;   // b*S + s
                int col = n0 + wn * 64 + j * 16 + nn;           // e in [0,3D)
                bf16 bv = f2bf(acc[i][j][r]);
                int part = col >> 10, rem = col & 1023;
                int h = rem >> 6, a = rem & 63;
                int b = row >> 11, s = row & 2047;
                size_t bh = (size_t)(b * NHH + h);
                if (part == 0)      Qb[(bh * SS + s) * AD + a] = bv;
                else if (part == 1) Kb[(bh * SS + s) * AD + a] = bv;
                else                Vtb[(bh * AD + a) * SS + s] = bv;
            }
}

// --------------------------------- kernel 3: relu^2-normalized attention
// per (b,h): scores = Q K^T / 8 + bias -> relu^2 -> P; O = (P V) / rowsum(P)
__global__ __launch_bounds__(256)
void attn_kernel(const bf16* __restrict__ Qb, const bf16* __restrict__ Kb,
                 const bf16* __restrict__ Vtb, bf16* __restrict__ O,
                 const float* __restrict__ sn_bias) {
    __shared__ bf16 Ks[64][72];        // [key][a]
    __shared__ bf16 Vs[64][72];        // [a][key]   (from V-transposed)
    __shared__ bf16 Ps[8][16][72];     // per-wave P slab: [m][key]
    const int tid = threadIdx.x, lane = tid & 31, w = tid >> 5;
    const int hl = lane >> 4, nn = lane & 15;
    const int h = blockIdx.y, b = blockIdx.z;
    const int q0 = blockIdx.x * 128 + w * 16;     // 16 q-rows per wave
    const size_t bh = (size_t)(b * NHH + h);
    const bf16* Qh  = Qb  + bh * SS * AD;
    const bf16* Kh  = Kb  + bh * SS * AD;
    const bf16* Vth = Vtb + bh * AD * SS;
    const float bias  = sn_bias[0];
    const float scale = 0.125f;                   // 1/sqrt(64)

    // Q fragments held in registers for the whole key loop (global b128 loads)
    v16bf qf[2];
    qf[0] = load_afrag(Qh + (size_t)q0 * AD, AD);
    qf[1] = load_afrag(Qh + (size_t)q0 * AD + 32, AD);

    v8f oacc[4];
    #pragma unroll
    for (int i = 0; i < 4; ++i) oacc[i] = vzero8();
    v8f rs = vzero8();

    for (int kb = 0; kb < SS / 64; ++kb) {
#if HAVE_TDM
        // TDM: DMA K tile (64 keys x 64) and Vt tile (64 x 64 keys) into LDS,
        // hardware-padded to a 72-bf16 row stride. Wave 0 issues + waits.
        if (w == 0) {
            tdm_load_2d((unsigned)(size_t)&Ks[0][0],
                        Kh + (size_t)(kb * 64) * AD,
                        /*tensor_d0=*/AD, /*tensor_d1=*/SS,
                        /*tile_d0=*/64, /*tile_d1=*/64, /*stride0=*/AD);
            tdm_load_2d((unsigned)(size_t)&Vs[0][0],
                        Vth + (size_t)(kb * 64),
                        /*tensor_d0=*/SS, /*tensor_d1=*/AD,
                        /*tile_d0=*/64, /*tile_d1=*/64, /*stride0=*/SS);
            __builtin_amdgcn_s_wait_tensorcnt(0);
        }
        __syncthreads();
#else
        // fallback: cooperative stage of K (64x64) and Vt (64x64)
        for (int r = tid >> 3; r < 64; r += 32) {
            int c = (tid & 7) * 8;
            *(v8bf*)&Ks[r][c] = *(const v8bf*)(Kh + (size_t)(kb * 64 + r) * AD + c);
            *(v8bf*)&Vs[r][c] = *(const v8bf*)(Vth + (size_t)r * SS + kb * 64 + c);
        }
        __syncthreads();
#endif
        // scores for 4 key sub-tiles, activation, P to LDS, rowsum partials
        #pragma unroll
        for (int nt = 0; nt < 4; ++nt) {
            v8f sc = vzero8();
            v16bf bk0 = load_bfrag(&Ks[nt * 16][0], 72);
            v16bf bk1 = load_bfrag(&Ks[nt * 16][32], 72);
            sc = wmma_bf16(qf[0], bk0, sc);
            sc = wmma_bf16(qf[1], bk1, sc);
            #pragma unroll
            for (int r = 0; r < 8; ++r) {
                float t = fmaxf(sc[r] * scale + bias, 0.0f);
                t = t * t;
                rs[r] += t;
                Ps[w][r + 8 * hl][nt * 16 + nn] = f2bf(t);
            }
        }
        // P (16x64) x V (64x64): A frags from own LDS slab (in-order DS)
        v16bf pa0 = load_afrag(&Ps[w][0][0], 72);
        v16bf pa1 = load_afrag(&Ps[w][0][32], 72);
        #pragma unroll
        for (int na = 0; na < 4; ++na) {
            v16bf bv0 = load_bfrag(&Vs[na * 16][0], 72);
            v16bf bv1 = load_bfrag(&Vs[na * 16][32], 72);
            oacc[na] = wmma_bf16(pa0, bv0, oacc[na]);
            oacc[na] = wmma_bf16(pa1, bv1, oacc[na]);
        }
        __syncthreads();
    }

    // rowsum: reduce partials across the 16 lanes of each half-group
    #pragma unroll
    for (int r = 0; r < 8; ++r) {
        float v = rs[r];
        v += __shfl_xor(v, 1, 32);
        v += __shfl_xor(v, 2, 32);
        v += __shfl_xor(v, 4, 32);
        v += __shfl_xor(v, 8, 32);
        rs[r] = v;
    }
    // normalize and store O as bf16 [b][s][h*64+a]
    #pragma unroll
    for (int na = 0; na < 4; ++na)
        #pragma unroll
        for (int r = 0; r < 8; ++r) {
            int m = r + 8 * hl;
            int s = q0 + m;
            float v = oacc[na][r] / (rs[r] + 1e-32f);
            O[((size_t)(b * SS + s)) * DD + h * AD + na * 16 + nn] = f2bf(v);
        }
}

// ----------------------------------------- kernel 4: output projection GEMM
// out:(B*S, D) f32 = A:(B*S, D) bf16  x  Wout^T (Wout:(D,D) bf16 row-major)
__global__ __launch_bounds__(256)
void out_gemm_kernel(const bf16* __restrict__ A, const bf16* __restrict__ W,
                     float* __restrict__ out) {
    __shared__ bf16 As[128][40];
    __shared__ bf16 Bs[128][40];
    const int tid = threadIdx.x, lane = tid & 31, w = tid >> 5;
    const int wm = w >> 1, wn = w & 1;
    const int m0 = blockIdx.x * 128;
    const int n0 = blockIdx.y * 128;
    const int hl = lane >> 4, nn = lane & 15;

    v8f acc[2][4];
    #pragma unroll
    for (int i = 0; i < 2; ++i)
        #pragma unroll
        for (int j = 0; j < 4; ++j) acc[i][j] = vzero8();

    for (int k0 = 0; k0 < DD; k0 += 32) {
        for (int r = tid >> 2; r < 128; r += 64) {
            int c = (tid & 3) * 8;
            const bf16* pa = A + (size_t)(m0 + r) * DD + k0 + c;
            const bf16* pw = W + (size_t)(n0 + r) * DD + k0 + c;
            *(v8bf*)&As[r][c] = *(const v8bf*)pa;
            *(v8bf*)&Bs[r][c] = *(const v8bf*)pw;
            if (k0 + 32 < DD) {
                __builtin_prefetch(pa + 32, 0, 1);
                __builtin_prefetch(pw + 32, 0, 1);
            }
        }
        __syncthreads();
        v16bf af[2], bfr[4];
        af[0] = load_afrag(&As[wm * 32 + 0][0], 40);
        af[1] = load_afrag(&As[wm * 32 + 16][0], 40);
        #pragma unroll
        for (int j = 0; j < 4; ++j) bfr[j] = load_bfrag(&Bs[wn * 64 + j * 16][0], 40);
        #pragma unroll
        for (int i = 0; i < 2; ++i)
            #pragma unroll
            for (int j = 0; j < 4; ++j)
                acc[i][j] = wmma_bf16(af[i], bfr[j], acc[i][j]);
        __syncthreads();
    }
    #pragma unroll
    for (int i = 0; i < 2; ++i)
        #pragma unroll
        for (int j = 0; j < 4; ++j)
            #pragma unroll
            for (int r = 0; r < 8; ++r) {
                int row = m0 + wm * 32 + i * 16 + r + 8 * hl;
                int col = n0 + wn * 64 + j * 16 + nn;
                out[(size_t)row * DD + col] = acc[i][j][r];
            }
}

// ------------------------------------------------------------------ launch
extern "C" void kernel_launch(void* const* d_in, const int* in_sizes, int n_in,
                              void* d_out, int out_size, void* d_ws, size_t ws_size,
                              hipStream_t stream) {
    const float* iQ      = (const float*)d_in[0];
    const float* w_qkv   = (const float*)d_in[1];
    const float* w_out   = (const float*)d_in[2];
    const float* sn_bias = (const float*)d_in[3];

    char* ws = (char*)d_ws;
    size_t off = 0;
    bf16* wqkv_b = (bf16*)(ws + off); off += (size_t)3 * DD * DD * sizeof(bf16); // 6 MB
    bf16* wout_b = (bf16*)(ws + off); off += (size_t)DD * DD * sizeof(bf16);     // 2 MB
    bf16* Qb     = (bf16*)(ws + off); off += (size_t)BB * NHH * SS * AD * sizeof(bf16); // 16 MB
    bf16* Kb     = (bf16*)(ws + off); off += (size_t)BB * NHH * SS * AD * sizeof(bf16); // 16 MB
    bf16* Vtb    = (bf16*)(ws + off); off += (size_t)BB * NHH * AD * SS * sizeof(bf16); // 16 MB
    bf16* Ob     = (bf16*)(ws + off); off += (size_t)BB * SS * DD * sizeof(bf16);       // 16 MB

    cvt_bf16_kernel<<<dim3(2048), dim3(256), 0, stream>>>(w_qkv, wqkv_b, 3 * DD * DD);
    cvt_bf16_kernel<<<dim3(1024), dim3(256), 0, stream>>>(w_out, wout_b, DD * DD);

    qkv_gemm_kernel<<<dim3((BB * SS) / 128, (3 * DD) / 128), dim3(256), 0, stream>>>(
        iQ, wqkv_b, Qb, Kb, Vtb);

    attn_kernel<<<dim3(SS / 128, NHH, BB), dim3(256), 0, stream>>>(
        Qb, Kb, Vtb, Ob, sn_bias);

    out_gemm_kernel<<<dim3((BB * SS) / 128, DD / 128), dim3(256), 0, stream>>>(
        Ob, wout_b, (float*)d_out);
}
